// GraphSAGEModel_21311627723482
// MI455X (gfx1250) — matrix operand order repaired and checked by reading, not compiled, not run on hardware
//
#include <hip/hip_runtime.h>
#include <hip/hip_bf16.h>

#define NN 50000
#define NE 800000
#define D 128
#define DOUT 64

typedef __attribute__((ext_vector_type(16))) _Float16 v16h;
typedef __attribute__((ext_vector_type(8)))  float    v8f;

union Frag { v16h h; unsigned int u[8]; };

// ---------------- small utility kernels ----------------

__global__ __launch_bounds__(256) void f32_to_f16_kernel(const float* __restrict__ in,
                                                         _Float16* __restrict__ out, int n) {
    int i = blockIdx.x * 256 + threadIdx.x;
    if (i < n) out[i] = (_Float16)in[i];
}

__global__ __launch_bounds__(256) void zero_f32_kernel(float* __restrict__ p, int n) {
    int i = blockIdx.x * 256 + threadIdx.x;
    if (i < n) p[i] = 0.0f;
}

// ---------------- edge scatter-sum (mean numerator + degree) ----------------
// blockDim = 128 (one thread per feature); each block handles 16 edges.
__global__ __launch_bounds__(128) void scatter_sum_kernel(const float* __restrict__ x,
                                                          const int* __restrict__ src,
                                                          const int* __restrict__ dst,
                                                          float* __restrict__ sums,
                                                          float* __restrict__ cnt) {
    const int t  = threadIdx.x;
    const int e0 = blockIdx.x * 16;
#pragma unroll 1
    for (int i = 0; i < 16; ++i) {
        int e = e0 + i;
        if (e >= NE) break;
        int s = src[e];
        int d = dst[e];
        atomicAdd(&sums[d * D + t], x[s * D + t]);
        if (t == 0) atomicAdd(&cnt[d], 1.0f);
    }
}

// ---------------- WMMA fragment helpers ----------------

// A fragment (16x32 f16, M x K) from an LDS tile stored row-major [16][128] halves.
__device__ inline Frag load_a_frag(const _Float16* lds, int m, int koff, int kbase) {
    Frag a;
#pragma unroll
    for (int j = 0; j < 8; ++j) {
        int k = kbase + (koff << 1) + 2 * j;
        a.u[j] = *(const unsigned int*)(&lds[(m << 7) + k]);
    }
    return a;
}

// B fragment (32x16 f16, K x N) for out = X @ W^T: B[k][n] = W[n][k], W row-major [n][128] f16.
__device__ inline Frag load_b_frag(const _Float16* __restrict__ W, int n, int lane, int kbase) {
    Frag b;
    const unsigned int* wp =
        (const unsigned int*)(W + (n << 7) + kbase + ((lane >> 4) << 4));
#pragma unroll
    for (int j = 0; j < 8; ++j) b.u[j] = wp[j];
    return b;
}

// ---------------- SAGE layer: out = relu(l2norm(x@Wl^T + bl + agg@Wr^T + br)) ----------------
// grid = 3125 blocks (16 nodes each), block = 256 threads = 8 waves; wave w owns cols [16w,16w+16).
__global__ __launch_bounds__(256) void sage_combine_kernel(const float* __restrict__ x,
                                                           const float* __restrict__ sums,
                                                           const float* __restrict__ cnt,
                                                           const _Float16* __restrict__ Wlh,
                                                           const float* __restrict__ bl,
                                                           const _Float16* __restrict__ Wrh,
                                                           const float* __restrict__ br,
                                                           float* __restrict__ out) {
    __shared__ _Float16 xh[16 * D];
    __shared__ _Float16 ah[16 * D];
    __shared__ float    ot[16 * D];
    __shared__ float    invn[16];

    const int tid  = threadIdx.x;
    const int row0 = blockIdx.x * 16;

    for (int i = tid; i < 16 * D; i += 256) {
        int mm = i >> 7, k = i & 127;
        int node = row0 + mm;
        xh[i] = (_Float16)x[node * D + k];
        float c = cnt[node];
        c = c > 1.0f ? c : 1.0f;
        ah[i] = (_Float16)(sums[node * D + k] / c);
    }
    __syncthreads();

    const int w    = tid >> 5;
    const int lane = tid & 31;
    const int m    = lane & 15;
    const int koff = (lane >> 4) << 3;
    const int nloc = (w << 4) + (lane & 15);

    v8f acc = {};
#pragma unroll
    for (int kc = 0; kc < 4; ++kc) {
        Frag a = load_a_frag(xh, m, koff, kc << 5);
        Frag b = load_b_frag(Wlh, nloc, lane, kc << 5);
        acc = __builtin_amdgcn_wmma_f32_16x16x32_f16(false, a.h, false, b.h,
                                                     (short)0, acc, false, false);
    }
#pragma unroll
    for (int kc = 0; kc < 4; ++kc) {
        Frag a = load_a_frag(ah, m, koff, kc << 5);
        Frag b = load_b_frag(Wrh, nloc, lane, kc << 5);
        acc = __builtin_amdgcn_wmma_f32_16x16x32_f16(false, a.h, false, b.h,
                                                     (short)0, acc, false, false);
    }

    const float bias = bl[nloc] + br[nloc];
    const int   mb   = (lane >> 4) << 3;
#pragma unroll
    for (int j = 0; j < 8; ++j) ot[((mb + j) << 7) + nloc] = acc[j] + bias;
    __syncthreads();

    if (tid < 16) {
        float s = 0.0f;
        for (int n = 0; n < D; ++n) {
            float v = ot[(tid << 7) + n];
            s += v * v;
        }
        float nrm = sqrtf(s);
        nrm = nrm > 1e-12f ? nrm : 1e-12f;
        invn[tid] = 1.0f / nrm;
    }
    __syncthreads();

    for (int i = tid; i < 16 * D; i += 256) {
        int mm = i >> 7;
        float v = ot[i] * invn[mm];
        out[(row0 + mm) * D + (i & 127)] = v > 0.0f ? v : 0.0f;
    }
}

// ---------------- post_mp: out = relu(x@Wp1^T + bp1) @ Wp2^T + bp2 ----------------
__global__ __launch_bounds__(256) void post_mp_kernel(const float* __restrict__ x,
                                                      const _Float16* __restrict__ Wp1h,
                                                      const float* __restrict__ bp1,
                                                      const _Float16* __restrict__ Wp2h,
                                                      const float* __restrict__ bp2,
                                                      float* __restrict__ out) {
    __shared__ _Float16 xh[16 * D];
    __shared__ _Float16 hh[16 * D];

    const int tid  = threadIdx.x;
    const int row0 = blockIdx.x * 16;

    for (int i = tid; i < 16 * D; i += 256) {
        int mm = i >> 7, k = i & 127;
        xh[i] = (_Float16)x[(row0 + mm) * D + k];
    }
    __syncthreads();

    const int w    = tid >> 5;
    const int lane = tid & 31;
    const int m    = lane & 15;
    const int koff = (lane >> 4) << 3;
    const int nloc = (w << 4) + (lane & 15);
    const int mb   = (lane >> 4) << 3;

    v8f acc = {};
#pragma unroll
    for (int kc = 0; kc < 4; ++kc) {
        Frag a = load_a_frag(xh, m, koff, kc << 5);
        Frag b = load_b_frag(Wp1h, nloc, lane, kc << 5);
        acc = __builtin_amdgcn_wmma_f32_16x16x32_f16(false, a.h, false, b.h,
                                                     (short)0, acc, false, false);
    }
    const float b1 = bp1[nloc];
#pragma unroll
    for (int j = 0; j < 8; ++j) {
        float v = acc[j] + b1;
        hh[((mb + j) << 7) + nloc] = (_Float16)(v > 0.0f ? v : 0.0f);
    }
    __syncthreads();

    if (w < 4) {  // 64 output columns -> 4 waves; wave-uniform branch, EXEC stays all-ones
        const int n2 = (w << 4) + (lane & 15);
        v8f acc2 = {};
#pragma unroll
        for (int kc = 0; kc < 4; ++kc) {
            Frag a = load_a_frag(hh, m, koff, kc << 5);
            Frag b = load_b_frag(Wp2h, n2, lane, kc << 5);
            acc2 = __builtin_amdgcn_wmma_f32_16x16x32_f16(false, a.h, false, b.h,
                                                          (short)0, acc2, false, false);
        }
        const float b2 = bp2[n2];
#pragma unroll
        for (int j = 0; j < 8; ++j) out[(row0 + mb + j) * DOUT + n2] = acc2[j] + b2;
    }
}

// ---------------- launcher ----------------

extern "C" void kernel_launch(void* const* d_in, const int* in_sizes, int n_in,
                              void* d_out, int out_size, void* d_ws, size_t ws_size,
                              hipStream_t stream) {
    const float* xin = (const float*)d_in[0];
    const float* Wl0 = (const float*)d_in[1];  const float* bl0 = (const float*)d_in[2];
    const float* Wr0 = (const float*)d_in[3];  const float* br0 = (const float*)d_in[4];
    const float* Wl1 = (const float*)d_in[5];  const float* bl1 = (const float*)d_in[6];
    const float* Wr1 = (const float*)d_in[7];  const float* br1 = (const float*)d_in[8];
    const float* Wp1 = (const float*)d_in[9];  const float* bp1 = (const float*)d_in[10];
    const float* Wp2 = (const float*)d_in[11]; const float* bp2 = (const float*)d_in[12];
    const int*   eix = (const int*)d_in[13];
    const int*   src = eix;
    const int*   dst = eix + NE;

    char* ws = (char*)d_ws;
    float*    sums = (float*)ws;                                   // 25,600,000 B
    float*    cnt  = (float*)(ws + (size_t)NN * D * 4);            //    200,000 B (contiguous)
    float*    h1   = (float*)(ws + 25800000);                      // 25,600,000 B
    float*    h2   = (float*)(ws + 51400000);                      // 25,600,000 B
    _Float16* wh   = (_Float16*)(ws + 77000000);
    _Float16* Wl0h = wh;              _Float16* Wr0h = wh + 16384;
    _Float16* Wl1h = wh + 32768;      _Float16* Wr1h = wh + 49152;
    _Float16* Wp1h = wh + 65536;      _Float16* Wp2h = wh + 81920;

    // Stage weights once as f16.
    const float* wsrc[6] = {Wl0, Wr0, Wl1, Wr1, Wp1, Wp2};
    _Float16*    wdst[6] = {Wl0h, Wr0h, Wl1h, Wr1h, Wp1h, Wp2h};
    const int    wn[6]   = {16384, 16384, 16384, 16384, 16384, 8192};
    for (int i = 0; i < 6; ++i)
        f32_to_f16_kernel<<<(wn[i] + 255) / 256, 256, 0, stream>>>(wsrc[i], wdst[i], wn[i]);

    const int nzero   = NN * D + NN;          // sums + cnt, contiguous
    const int zgrid   = (nzero + 255) / 256;
    const int egrid   = (NE + 15) / 16;       // 16 edges per block
    const int rowgrid = NN / 16;              // 3125, exact

    // Layer 0
    zero_f32_kernel<<<zgrid, 256, 0, stream>>>(sums, nzero);
    scatter_sum_kernel<<<egrid, 128, 0, stream>>>(xin, src, dst, sums, cnt);
    sage_combine_kernel<<<rowgrid, 256, 0, stream>>>(xin, sums, cnt, Wl0h, bl0, Wr0h, br0, h1);

    // Layer 1
    zero_f32_kernel<<<zgrid, 256, 0, stream>>>(sums, nzero);
    scatter_sum_kernel<<<egrid, 128, 0, stream>>>(h1, src, dst, sums, cnt);
    sage_combine_kernel<<<rowgrid, 256, 0, stream>>>(h1, sums, cnt, Wl1h, bl1, Wr1h, br1, h2);

    // post_mp
    post_mp_kernel<<<rowgrid, 256, 0, stream>>>(h2, Wp1h, bp1, Wp2h, bp2, (float*)d_out);
}
